// AudioPreprocessingLayer_39341900431994
// MI455X (gfx1250) — compile-verified, matrix operand order checked by reading, not run on hardware
//
#include <hip/hip_runtime.h>
#include <hip/hip_bf16.h>
#include <math.h>

// ---------------------------------------------------------------------------
// Audio preprocessing as two WMMA GEMMs:
//   R   = X (131072x512)  @ D^T (512x272)      D[f][n] = hw[n]*cos(2*pi*f*n/512)/512
//   MAG = R*R * 2^20                           (scale keeps f16 in normal range)
//   MEL = MAG (131072x288) @ FB^T (288x32)     (freqs padded 272->288, mels 20->32)
//   out = (MEL==0) ? -52 : floor(log2(MEL)) - 20
// B panels of the cosine matrix are streamed into LDS with the CDNA5 async
// path (global_load_async_to_lds_b128, ASYNCcnt), double-buffered, and shared
// by all 8 waves of the workgroup.
// ---------------------------------------------------------------------------

typedef _Float16 v16h __attribute__((ext_vector_type(16)));
typedef _Float16 v8h  __attribute__((ext_vector_type(8)));
typedef float    v8f  __attribute__((ext_vector_type(8)));

#define NFRAMES    131072   // 4096 * 32
#define K1         512      // samples per frame
#define NT1        17       // 257 freqs padded to 17 tiles of 16
#define KT1        16       // 512 / 32
#define NF2        288      // stage-2 K: 272 padded to 9 tiles of 32
#define KT2        9
#define NT2        2        // 20 mels padded to 32
#define WAVES      8
#define BLOCK      256
#define FR_PER_WG  128      // WAVES * 16
#define PANEL_H    (KT1 * 32 * 16)          // 8192 f16 = 16 KiB per nt panel
#define W1_ELEMS   (NT1 * PANEL_H)          // 139264 f16
#define W2_ELEMS   (NT2 * KT2 * 32 * 16)    // 9216 f16
#define LDS_A      (WAVES * 16 * K1)        // 65536 f16
#define LDS_M      (WAVES * 16 * NF2)       // 36864 f16
#define LDS_BYTES  ((LDS_A + LDS_M + 2 * PANEL_H) * 2)   // 232 KiB

// Per-lane K index inside a 16-bit 16x32 A/B fragment (CDNA5 ISA 7.12.2):
// lanes 0-15 hold K {0..7,16..23}; lanes 16-31 hold K {8..15,24..31}.
__device__ __forceinline__ int kmap(int lane, int h) {
    return ((lane >> 4) << 3) + (h & 7) + ((h >> 3) << 4);
}

// Raw LDS byte offset of a generic pointer into shared memory.
__device__ __forceinline__ uint32_t lds_off(const void* p) {
    return (uint32_t)(uintptr_t)(__attribute__((address_space(3))) const void*)p;
}

// CDNA5 async copy: 16 bytes global -> LDS, tracked by ASYNCcnt.
// VGLOBAL GVS form: vdst = LDS byte address, vaddr = u32 offset, saddr = u64 base.
__device__ __forceinline__ void async_copy_b128(uint32_t lds_byte, uint32_t voff,
                                                uint64_t gbase) {
    asm volatile("global_load_async_to_lds_b128 %0, %1, %2"
                 :: "v"(lds_byte), "v"(voff), "s"(gbase)
                 : "memory");
}
__device__ __forceinline__ void wait_asynccnt0() {
    asm volatile("s_wait_asynccnt 0x0" ::: "memory");
}

// Load a 16x32 f16 fragment from a row-major f16 array in LDS: per lane, two
// contiguous 16-byte ds loads at K = kOff+kb and K = kOff+kb+16.
__device__ __forceinline__ v16h load_frag_rows(const _Float16* base, int rowStride,
                                               int lane, int kOff) {
    const int row = lane & 15;
    const int kb  = (lane >> 4) << 3;
    const _Float16* p = base + row * rowStride + kOff + kb;
    v8h lo = *(const v8h*)(p);
    v8h hi = *(const v8h*)(p + 16);
    v16h r;
#pragma unroll
    for (int i = 0; i < 8; ++i) { r[i] = lo[i]; r[8 + i] = hi[i]; }
    return r;
}

// Pre-swizzled B fragment: one 32-byte contiguous read per lane.
__device__ __forceinline__ v16h load_frag_sw(const _Float16* w, int fragIdx, int lane) {
    return *(const v16h*)(w + ((size_t)fragIdx * 32 + lane) * 16);
}

// Issue async load of W1 panel `nt` (16 KiB) into LDS buffer, spread over block.
__device__ __forceinline__ void issue_panel(const _Float16* __restrict__ w1, int nt,
                                            _Float16* bpBuf, int tid) {
    const uint64_t gbase = (uint64_t)(uintptr_t)(w1 + (size_t)nt * PANEL_H);
    const uint32_t lbase = lds_off(bpBuf);
#pragma unroll
    for (int t = 0; t < 4; ++t) {
        uint32_t idx = (uint32_t)tid + (uint32_t)t * BLOCK;   // 0..1023 chunks
        async_copy_b128(lbase + idx * 16u, idx * 16u, gbase);
    }
}

// ---- prep: window-folded cosine matrix, pre-swizzled B layout ----
__global__ void build_w1(const float* __restrict__ hw, _Float16* __restrict__ w1) {
    int idx = blockIdx.x * 256 + threadIdx.x;
    if (idx >= W1_ELEMS) return;
    int h    = idx & 15;
    int lane = (idx >> 4) & 31;
    int rest = idx >> 9;           // nt*16 + kt
    int kt   = rest & 15;
    int nt   = rest >> 4;
    int f    = nt * 16 + (lane & 15);
    int n    = kt * 32 + kmap(lane, h);
    float v  = 0.0f;
    if (f < 257) {
        int ph = (f * n) & 511;    // exact phase reduction: cos(2*pi*ph/512)
        v = hw[n] * __cosf((float)ph * 0.012271846303085129f) * (1.0f / 512.0f);
    }
    w1[idx] = (_Float16)v;
}

// ---- prep: padded mel filterbank transpose, pre-swizzled B layout ----
__global__ void build_w2(const float* __restrict__ fb, _Float16* __restrict__ w2) {
    int idx = blockIdx.x * 256 + threadIdx.x;
    if (idx >= W2_ELEMS) return;
    int h    = idx & 15;
    int lane = (idx >> 4) & 31;
    int rest = idx >> 9;           // nt*9 + kt
    int kt   = rest % 9;
    int nt   = rest / 9;
    int m    = nt * 16 + (lane & 15);
    int f    = kt * 32 + kmap(lane, h);
    float v  = (m < 20 && f < 257) ? fb[m * 257 + f] : 0.0f;
    w2[idx] = (_Float16)v;
}

// ---- main: 8 waves x 16 frames; async-staged shared B panels ----
__global__ void __launch_bounds__(BLOCK, 1)
mel_log2_kernel(const float* __restrict__ x, const _Float16* __restrict__ w1,
                const _Float16* __restrict__ w2, float* __restrict__ out) {
    extern __shared__ char smem[];
    const int tid  = threadIdx.x;
    const int lane = tid & 31;
    const int wave = tid >> 5;
    const int frameBase = blockIdx.x * FR_PER_WG + wave * 16;

    _Float16* Alds = (_Float16*)smem;          // [8][16][512]
    _Float16* Mlds = Alds + LDS_A;             // [8][16][288]
    _Float16* Bp0  = Mlds + LDS_M;             // [16][32][16] panel, double-buffered
    _Float16* Bp1  = Bp0 + PANEL_H;
    _Float16* AldsW = Alds + (size_t)wave * (16 * K1);
    _Float16* MldsW = Mlds + (size_t)wave * (16 * NF2);

    // Kick off async DMA of the first B panel, overlap with input staging.
    issue_panel(w1, 0, Bp0, tid);

    // Stage this wave's 16 frames (f32 -> f16) into LDS; zero padded mag buffer.
    const float* xw = x + (size_t)frameBase * K1;
#pragma unroll 4
    for (int i = lane; i < 16 * K1; i += 32) AldsW[i] = (_Float16)xw[i];
    for (int i = lane; i < 16 * NF2; i += 32) MldsW[i] = (_Float16)0.0f;

    wait_asynccnt0();
    __syncthreads();

    // Stage 1: R = X @ D^T with double-buffered async B panels,
    // then mag = (R^2) * 2^20 back to LDS.
    for (int nt = 0; nt < NT1; ++nt) {
        const _Float16* Bcur = (nt & 1) ? Bp1 : Bp0;
        _Float16*       Bnxt = (nt & 1) ? Bp0 : Bp1;
        if (nt + 1 < NT1) issue_panel(w1, nt + 1, Bnxt, tid);

        v8f acc = {};
#pragma unroll
        for (int kt = 0; kt < KT1; ++kt) {
            v16h a = load_frag_rows(AldsW, K1, lane, kt * 32);
            v16h b = *(const v16h*)(Bcur + ((kt * 32 + lane) << 4));
            acc = __builtin_amdgcn_wmma_f32_16x16x32_f16(
                false, a, false, b, (short)0, acc, false, false);
        }
        const int f     = nt * 16 + (lane & 15);
        const int rbase = (lane >> 4) << 3;
#pragma unroll
        for (int r = 0; r < 8; ++r) {
            float v = acc[r];
            MldsW[(rbase + r) * NF2 + f] = (_Float16)(v * v * 1048576.0f);
        }
        wait_asynccnt0();
        __syncthreads();
    }

    // Stage 2: MEL = MAG @ FB^T, then floor(log2) - 20.
    for (int nt = 0; nt < NT2; ++nt) {
        v8f acc = {};
#pragma unroll
        for (int kt = 0; kt < KT2; ++kt) {
            v16h a = load_frag_rows(MldsW, NF2, lane, kt * 32);
            v16h b = load_frag_sw(w2, nt * KT2 + kt, lane);
            acc = __builtin_amdgcn_wmma_f32_16x16x32_f16(
                false, a, false, b, (short)0, acc, false, false);
        }
        const int mel = nt * 16 + (lane & 15);
        if (mel < 20) {
            const int rbase = (lane >> 4) << 3;
#pragma unroll
            for (int r = 0; r < 8; ++r) {
                float c = acc[r];
                float res = (c <= 0.0f) ? -52.0f : (floorf(__log2f(c)) - 20.0f);
                out[(size_t)(frameBase + rbase + r) * 20 + mel] = res;
            }
        }
    }
}

extern "C" void kernel_launch(void* const* d_in, const int* in_sizes, int n_in,
                              void* d_out, int out_size, void* d_ws, size_t ws_size,
                              hipStream_t stream) {
    const float* x  = (const float*)d_in[0];  // (4096, 32, 512)
    const float* fb = (const float*)d_in[1];  // (20, 257)
    const float* hw = (const float*)d_in[2];  // (512,)
    float* out = (float*)d_out;               // (4096, 32, 20, 1)

    _Float16* w1 = (_Float16*)d_ws;                                   // 278528 B
    _Float16* w2 = (_Float16*)((char*)d_ws + (size_t)W1_ELEMS * 2);   // 18432 B

    build_w1<<<(W1_ELEMS + 255) / 256, 256, 0, stream>>>(hw, w1);
    build_w2<<<(W2_ELEMS + 255) / 256, 256, 0, stream>>>(fb, w2);

    const int numWG = NFRAMES / FR_PER_WG;  // 1024, exact tiling -> full EXEC
    mel_log2_kernel<<<numWG, BLOCK, LDS_BYTES, stream>>>(x, w1, w2, out);
}
